// WindowAttention_66425964200053
// MI455X (gfx1250) — matrix operand order repaired and checked by reading, not compiled, not run on hardware
//
#include <hip/hip_runtime.h>
#include <hip/hip_bf16.h>

// ---------------------------------------------------------------------------
// Windowed MHA (T=4096, D=1024, H=16, hd=64, WINDOW=128) for MI455X/gfx1250.
// - all matmuls via v_wmma_f32_16x16x32_f16 (wave32), fp32 accumulation
// - K/A tiles staged in LDS with GLOBAL_LOAD_ASYNC_TO_LDS_B128, double-buffered
// - V stored pre-transposed so PV B-fragments are contiguous global loads
// ---------------------------------------------------------------------------

typedef __attribute__((ext_vector_type(16))) _Float16 v16h;
typedef __attribute__((ext_vector_type(8)))  _Float16 v8h;
typedef __attribute__((ext_vector_type(4)))  _Float16 v4h;
typedef __attribute__((ext_vector_type(8)))  float    v8f;

#define T_SEQ   4096
#define DMODEL  1024
#define NHEAD   16
#define HD      64
#define WIN     128

__device__ __forceinline__ v8f zero8() {
  v8f z = {0.f, 0.f, 0.f, 0.f, 0.f, 0.f, 0.f, 0.f};
  return z;
}

// ---- CDNA5 async copy: global -> LDS, 16B per lane, ASYNCcnt-tracked -------
__device__ __forceinline__ void async_ld16B(const _Float16* gptr, const _Float16* lptr) {
  unsigned lds_off = (unsigned)(uintptr_t)lptr;        // low 32b of generic = LDS offset
  unsigned long long ga = (unsigned long long)(uintptr_t)gptr;
  asm volatile("global_load_async_to_lds_b128 %0, %1, off"
               :: "v"(lds_off), "v"(ga) : "memory");
}

__device__ __forceinline__ void wait_async0() {
#if __has_builtin(__builtin_amdgcn_s_wait_asynccnt)
  __builtin_amdgcn_s_wait_asynccnt(0);
#else
  asm volatile("s_wait_asynccnt 0x0" ::: "memory");
#endif
}

// ---- fragment loaders (layouts per CDNA5 ISA §7.12.2) -----------
__device__ __forceinline__ v16h ld16h(const _Float16* p) {
  v8h lo = *(const v8h*)p;
  v8h hi = *(const v8h*)(p + 8);
  v16h o;
#pragma unroll
  for (int i = 0; i < 8; ++i) { o[i] = lo[i]; o[i + 8] = hi[i]; }
  return o;
}

// A-frag (16x32): lane holds row m=lane&15; half i -> k=(i>>3)*16+(lane>>4)*8+(i&7)
__device__ __forceinline__ v16h load_a_frag(const _Float16* tile, int ld) {
  const int lane = threadIdx.x & 31;
  const _Float16* r = tile + (size_t)(lane & 15) * ld + ((lane >> 4) << 3);
  v8h lo = *(const v8h*)r;
  v8h hi = *(const v8h*)(r + 16);
  v16h o;
#pragma unroll
  for (int i = 0; i < 8; ++i) { o[i] = lo[i]; o[i + 8] = hi[i]; }
  return o;
}

// B-frag (32x16) where B[k][n] = src[n][k]: lane holds col n=lane&15; half i -> k=(i>>3)*16+(lane>>4)*8+(i&7)
__device__ __forceinline__ v16h load_b_frag_nt(const _Float16* tile, size_t ld) {
  const int lane = threadIdx.x & 31;
  const _Float16* r = tile + (size_t)(lane & 15) * ld + ((lane >> 4) << 3);
  v8h lo = *(const v8h*)r;
  v8h hi = *(const v8h*)(r + 16);
  v16h o;
#pragma unroll
  for (int i = 0; i < 8; ++i) { o[i] = lo[i]; o[i + 8] = hi[i]; }
  return o;
}

__device__ __forceinline__ v8f wmma16(v16h a, v16h b, v8f c) {
  return __builtin_amdgcn_wmma_f32_16x16x32_f16(false, a, false, b,
                                                (short)0, c, false, false);
}

// ---------------------------------------------------------------------------
// fp32 -> fp16 conversion
// ---------------------------------------------------------------------------
__global__ void __launch_bounds__(256)
cvt_f32_f16(const float* __restrict__ in, _Float16* __restrict__ out, int n4) {
  int i = blockIdx.x * 256 + threadIdx.x;
  if (i < n4) {
    float4 v = ((const float4*)in)[i];
    v4h h;
    h[0] = (_Float16)v.x; h[1] = (_Float16)v.y;
    h[2] = (_Float16)v.z; h[3] = (_Float16)v.w;
    ((v4h*)out)[i] = h;
  }
}

// ---------------------------------------------------------------------------
// QKV projection NT-GEMM (4096 x 3072 x 1024), WG = 4 waves = 64x64 C tile.
// A (x) tile 64x32 is shared by all 4 waves: async-staged in LDS, double buf.
// Epilogue: q scaled by 1/8 into [H][T][64]; k into [H][T][64];
//           v written TRANSPOSED into [H][64][T] (rows of C are consecutive t).
// ---------------------------------------------------------------------------
__global__ void __launch_bounds__(128)
qkv_wmma(const _Float16* __restrict__ xh, const _Float16* __restrict__ w_inh,
         const float* __restrict__ b_in, _Float16* __restrict__ qh,
         _Float16* __restrict__ kh, _Float16* __restrict__ vT) {
  __shared__ __align__(16) _Float16 a_lds[2][64 * 40];   // [row][k], pad->40

  const int lane = threadIdx.x & 31;
  const int wave = threadIdx.x >> 5;
  const int m0 = blockIdx.x * 64;
  const int n0 = blockIdx.y * 64 + wave * 16;

  // coop async-load mapping: thread t -> row t>>1, 16-half chunk (t&1)*16
  const int arow = threadIdx.x >> 1;
  const int acol = (threadIdx.x & 1) << 4;
  const _Float16* asrc = xh + (size_t)(m0 + arow) * DMODEL + acol;

  v8f acc[4];
#pragma unroll
  for (int mt = 0; mt < 4; ++mt) acc[mt] = zero8();

  const _Float16* wrow = w_inh + (size_t)n0 * DMODEL;

  // prime buffer 0 with k0 = 0
  async_ld16B(asrc + 0, &a_lds[0][arow * 40 + acol]);
  async_ld16B(asrc + 8, &a_lds[0][arow * 40 + acol + 8]);

  for (int ks = 0; ks < DMODEL / 32; ++ks) {
    const int k0 = ks * 32;
    wait_async0();
    __syncthreads();                      // tile ready; prev buffer free
    if (ks + 1 < DMODEL / 32) {
      const _Float16* nsrc = asrc + k0 + 32;
      _Float16* nbuf = &a_lds[(ks + 1) & 1][arow * 40 + acol];
      async_ld16B(nsrc + 0, nbuf);
      async_ld16B(nsrc + 8, nbuf + 8);
    }
    if (k0 + 32 < DMODEL)
      __builtin_prefetch(wrow + (size_t)(lane & 15) * DMODEL + k0 + 32, 0, 1);
    v16h b = load_b_frag_nt(wrow + k0, DMODEL);
    const _Float16* abuf = &a_lds[ks & 1][0];
#pragma unroll
    for (int mt = 0; mt < 4; ++mt) {
      v16h a = load_a_frag(abuf + (mt * 16) * 40, 40);
      acc[mt] = wmma16(a, b, acc[mt]);
    }
  }

  // epilogue (n0 16-aligned: tile never crosses q/k/v or head boundaries)
  const int which = n0 >> 10;
  const int head  = (n0 & 1023) >> 6;
  const int colb  = n0 & 63;
  const int g  = lane >> 4;
  const int nn = lane & 15;
  const int col = colb + nn;
  const float bias = b_in[n0 + nn];

  if (which == 2) {
    // v: transposed store vT[head][col][t]; rows r are consecutive t -> v8h pack
#pragma unroll
    for (int mt = 0; mt < 4; ++mt) {
      v8h pk;
#pragma unroll
      for (int r = 0; r < 8; ++r) pk[r] = (_Float16)(acc[mt][r] + bias);
      int t0 = m0 + mt * 16 + g * 8;
      *(v8h*)&vT[((size_t)head * HD + col) * T_SEQ + t0] = pk;
    }
  } else {
    _Float16* dst = (which == 0) ? qh : kh;
    const float qs = (which == 0) ? 0.125f : 1.0f;  // fold 1/sqrt(hd) into q
#pragma unroll
    for (int mt = 0; mt < 4; ++mt) {
#pragma unroll
      for (int r = 0; r < 8; ++r) {
        int t = m0 + mt * 16 + r + g * 8;
        dst[((size_t)head * T_SEQ + t) * HD + col] =
            (_Float16)((acc[mt][r] + bias) * qs);
      }
    }
  }
}

// ---------------------------------------------------------------------------
// Windowed flash attention. Grid (T/64, H), 4 waves/WG.
// K blocks (32x64) async-staged in LDS (shared by 4 waves), double-buffered.
// V fragments read directly from the pre-transposed global vT.
// ---------------------------------------------------------------------------
__global__ void __launch_bounds__(128)
attn_wmma(const _Float16* __restrict__ qg, const _Float16* __restrict__ kg,
          const _Float16* __restrict__ vT, _Float16* __restrict__ ctxh) {
  __shared__ __align__(16) _Float16 k_lds[2][32 * 72];   // [key][d], pad->72
  __shared__ __align__(16) _Float16 p_lds[4][16 * 40];   // per-wave P tile

  const int lane = threadIdx.x & 31;
  const int wave = threadIdx.x >> 5;
  const int g    = lane >> 4;
  const int nn   = lane & 15;
  const int h    = blockIdx.y;
  const int qb   = blockIdx.x * 64;
  const int qrow0 = qb + wave * 16;

  const _Float16* Qh  = qg + (size_t)h * T_SEQ * HD;
  const _Float16* Kh  = kg + (size_t)h * T_SEQ * HD;
  const _Float16* VTh = vT + (size_t)h * HD * T_SEQ;

  // Q tile 16x64 as two resident A-fragments (already pre-scaled by 1/8)
  v16h qa0 = load_a_frag(Qh + (size_t)qrow0 * HD + 0, HD);
  v16h qa1 = load_a_frag(Qh + (size_t)qrow0 * HD + 32, HD);

  v8f o0 = zero8(), o1 = zero8(), o2 = zero8(), o3 = zero8();
  float m_r[8], l_r[8];
#pragma unroll
  for (int r = 0; r < 8; ++r) { m_r[r] = -1e30f; l_r[r] = 0.f; }

  const int kstart = (qb - WIN) > 0 ? (qb - WIN) : 0;           // 32-aligned
  const int kend   = (qb + 64 + WIN) < T_SEQ ? (qb + 64 + WIN) : T_SEQ;
  const int nblk   = (kend - kstart) >> 5;

  // coop async-load mapping for a 32x64 K block: thread t -> row t>>2, chunk (t&3)*16
  const int lrow = threadIdx.x >> 2;
  const int lcol = (threadIdx.x & 3) << 4;

  {
    const _Float16* src = Kh + (size_t)(kstart + lrow) * HD + lcol;
    async_ld16B(src + 0, &k_lds[0][lrow * 72 + lcol]);
    async_ld16B(src + 8, &k_lds[0][lrow * 72 + lcol + 8]);
  }

  for (int ib = 0; ib < nblk; ++ib) {
    const int kb = kstart + ib * 32;
    wait_async0();
    __syncthreads();                     // K block ready; other buffer free
    if (ib + 1 < nblk) {
      const _Float16* src = Kh + (size_t)(kb + 32 + lrow) * HD + lcol;
      _Float16* nbuf = &k_lds[(ib + 1) & 1][lrow * 72 + lcol];
      async_ld16B(src + 0, nbuf);
      async_ld16B(src + 8, nbuf + 8);
    }
    const _Float16* kbuf = &k_lds[ib & 1][0];

    // S = Q K^T (pre-scaled): two 16x16 tiles, each chained over d=64
    v8f s[2];
#pragma unroll
    for (int j = 0; j < 2; ++j) {
      v8f c = zero8();
      c = wmma16(qa0, load_b_frag_nt(kbuf + (j * 16) * 72 + 0,  72), c);
      c = wmma16(qa1, load_b_frag_nt(kbuf + (j * 16) * 72 + 32, 72), c);
      s[j] = c;
    }

    // window mask (element (row=r+g*8, col=nn))
#pragma unroll
    for (int j = 0; j < 2; ++j) {
#pragma unroll
      for (int r = 0; r < 8; ++r) {
        int d = (qrow0 + r + g * 8) - (kb + j * 16 + nn);
        bool ok = (d <= WIN) && (d >= -WIN);
        s[j][r] = ok ? s[j][r] : -1e30f;
      }
    }

    // online softmax: rows 0..7 in lanes 0..15, rows 8..15 in lanes 16..31;
    // width-16 xor butterflies reduce both halves simultaneously.
    float pr0[8], pr1[8];
#pragma unroll
    for (int r = 0; r < 8; ++r) {
      float bm = fmaxf(s[0][r], s[1][r]);
#pragma unroll
      for (int msk = 8; msk >= 1; msk >>= 1)
        bm = fmaxf(bm, __shfl_xor(bm, msk, 32));
      float mnew  = fmaxf(m_r[r], bm);
      float alpha = __expf(m_r[r] - mnew);
      float p0 = (s[0][r] > -1e29f) ? __expf(s[0][r] - mnew) : 0.f;
      float p1 = (s[1][r] > -1e29f) ? __expf(s[1][r] - mnew) : 0.f;
      float ps = p0 + p1;
#pragma unroll
      for (int msk = 8; msk >= 1; msk >>= 1)
        ps += __shfl_xor(ps, msk, 32);
      l_r[r] = l_r[r] * alpha + ps;
      m_r[r] = mnew;
      o0[r] = o0[r] * alpha; o1[r] = o1[r] * alpha;
      o2[r] = o2[r] * alpha; o3[r] = o3[r] * alpha;
      pr0[r] = p0; pr1[r] = p1;
    }

    // C-frag -> A-frag relayout of P via per-wave LDS (same-wave DS in-order)
    _Float16* pl = &p_lds[wave][0];
#pragma unroll
    for (int r = 0; r < 8; ++r) {
      pl[(r + g * 8) * 40 + 0  + nn] = (_Float16)pr0[r];
      pl[(r + g * 8) * 40 + 16 + nn] = (_Float16)pr1[r];
    }
    v16h pa = load_a_frag(pl, 40);       // P as 16x32 A-matrix

    // O += P * V : B-frags contiguous from pre-transposed vT[dim][t]
    o0 = wmma16(pa, load_b_frag_nt(VTh + (size_t)( 0) * T_SEQ + kb, T_SEQ), o0);
    o1 = wmma16(pa, load_b_frag_nt(VTh + (size_t)(16) * T_SEQ + kb, T_SEQ), o1);
    o2 = wmma16(pa, load_b_frag_nt(VTh + (size_t)(32) * T_SEQ + kb, T_SEQ), o2);
    o3 = wmma16(pa, load_b_frag_nt(VTh + (size_t)(48) * T_SEQ + kb, T_SEQ), o3);
  }

  // normalize and write ctx in [T][D] fp16 (heads re-interleaved)
#pragma unroll
  for (int r = 0; r < 8; ++r) {
    float inv = 1.f / l_r[r];
    int t = qrow0 + r + g * 8;
    size_t base = (size_t)t * DMODEL + h * HD + nn;
    ctxh[base + 0]  = (_Float16)(o0[r] * inv);
    ctxh[base + 16] = (_Float16)(o1[r] * inv);
    ctxh[base + 32] = (_Float16)(o2[r] * inv);
    ctxh[base + 48] = (_Float16)(o3[r] * inv);
  }
}

// ---------------------------------------------------------------------------
// Output projection + bias + residual. Same async A-tile staging as QKV.
// ---------------------------------------------------------------------------
__global__ void __launch_bounds__(128)
outproj_wmma(const _Float16* __restrict__ ctxh, const _Float16* __restrict__ w_outh,
             const float* __restrict__ b_out, const float* __restrict__ x,
             float* __restrict__ out) {
  __shared__ __align__(16) _Float16 a_lds[2][64 * 40];

  const int lane = threadIdx.x & 31;
  const int wave = threadIdx.x >> 5;
  const int m0 = blockIdx.x * 64;
  const int n0 = blockIdx.y * 64 + wave * 16;

  const int arow = threadIdx.x >> 1;
  const int acol = (threadIdx.x & 1) << 4;
  const _Float16* asrc = ctxh + (size_t)(m0 + arow) * DMODEL + acol;

  v8f acc[4];
#pragma unroll
  for (int mt = 0; mt < 4; ++mt) acc[mt] = zero8();

  const _Float16* wrow = w_outh + (size_t)n0 * DMODEL;

  async_ld16B(asrc + 0, &a_lds[0][arow * 40 + acol]);
  async_ld16B(asrc + 8, &a_lds[0][arow * 40 + acol + 8]);

  for (int ks = 0; ks < DMODEL / 32; ++ks) {
    const int k0 = ks * 32;
    wait_async0();
    __syncthreads();
    if (ks + 1 < DMODEL / 32) {
      const _Float16* nsrc = asrc + k0 + 32;
      _Float16* nbuf = &a_lds[(ks + 1) & 1][arow * 40 + acol];
      async_ld16B(nsrc + 0, nbuf);
      async_ld16B(nsrc + 8, nbuf + 8);
    }
    if (k0 + 32 < DMODEL)
      __builtin_prefetch(wrow + (size_t)(lane & 15) * DMODEL + k0 + 32, 0, 1);
    v16h b = load_b_frag_nt(wrow + k0, DMODEL);
    const _Float16* abuf = &a_lds[ks & 1][0];
#pragma unroll
    for (int mt = 0; mt < 4; ++mt) {
      v16h a = load_a_frag(abuf + (mt * 16) * 40, 40);
      acc[mt] = wmma16(a, b, acc[mt]);
    }
  }

  const int g  = lane >> 4;
  const int nn = lane & 15;
  const float bias = b_out[n0 + nn];
#pragma unroll
  for (int mt = 0; mt < 4; ++mt) {
#pragma unroll
    for (int r = 0; r < 8; ++r) {
      int t = m0 + mt * 16 + r + g * 8;
      size_t idx = (size_t)t * DMODEL + n0 + nn;
      out[idx] = x[idx] + acc[mt][r] + bias;
    }
  }
}

// ---------------------------------------------------------------------------
extern "C" void kernel_launch(void* const* d_in, const int* in_sizes, int n_in,
                              void* d_out, int out_size, void* d_ws, size_t ws_size,
                              hipStream_t stream) {
  const float* x     = (const float*)d_in[0];  // [T, D]
  const float* w_in  = (const float*)d_in[1];  // [3D, D]
  const float* b_in  = (const float*)d_in[2];  // [3D]
  const float* w_out = (const float*)d_in[3];  // [D, D]
  const float* b_out = (const float*)d_in[4];  // [D]
  float* out = (float*)d_out;

  char* ws = (char*)d_ws;
  size_t off = 0;
  _Float16* xh     = (_Float16*)(ws + off); off += (size_t)T_SEQ * DMODEL * 2;
  _Float16* w_inh  = (_Float16*)(ws + off); off += (size_t)3 * DMODEL * DMODEL * 2;
  _Float16* w_outh = (_Float16*)(ws + off); off += (size_t)DMODEL * DMODEL * 2;
  _Float16* qh     = (_Float16*)(ws + off); off += (size_t)NHEAD * T_SEQ * HD * 2;
  _Float16* kh     = (_Float16*)(ws + off); off += (size_t)NHEAD * T_SEQ * HD * 2;
  _Float16* vT     = (_Float16*)(ws + off); off += (size_t)NHEAD * HD * T_SEQ * 2;
  _Float16* ctxh   = (_Float16*)(ws + off); off += (size_t)T_SEQ * DMODEL * 2;
  (void)ws_size; (void)in_sizes; (void)n_in; (void)out_size;

  // fp32 -> fp16 staging
  cvt_f32_f16<<<T_SEQ * DMODEL / 4 / 256, 256, 0, stream>>>(x, xh, T_SEQ * DMODEL / 4);
  cvt_f32_f16<<<3 * DMODEL * DMODEL / 4 / 256, 256, 0, stream>>>(w_in, w_inh, 3 * DMODEL * DMODEL / 4);
  cvt_f32_f16<<<DMODEL * DMODEL / 4 / 256, 256, 0, stream>>>(w_out, w_outh, DMODEL * DMODEL / 4);

  // QKV projection (4096 x 3072 x 1024); q pre-scaled, v pre-transposed
  qkv_wmma<<<dim3(T_SEQ / 64, 3 * DMODEL / 64), 128, 0, stream>>>(
      xh, w_inh, b_in, qh, kh, vT);

  // windowed flash attention
  attn_wmma<<<dim3(T_SEQ / 64, NHEAD), 128, 0, stream>>>(qh, kh, vT, ctxh);

  // output projection + residual (4096 x 1024 x 1024)
  outproj_wmma<<<dim3(T_SEQ / 64, DMODEL / 64), 128, 0, stream>>>(
      ctxh, w_outh, b_out, x, out);
}
